// RNN_FeatureAttention_37340445671484
// MI455X (gfx1250) — compile-verified, matrix operand order checked by reading, not run on hardware
//
#include <hip/hip_runtime.h>
#include <hip/hip_bf16.h>

// ---- problem constants ----
#define BD    256       // threads per block (8 wave32)
#define NB    2         // batch elements per block
#define HT    24        // feature tokens
#define DM    128       // d_model
#define NHH   4         // heads
#define HDD   32        // head dim
#define FFH   512       // ffn hidden
#define TT    64        // time steps
#define MROWS (NB*HT)   // 48 rows = 3 WMMA m-tiles

typedef __attribute__((ext_vector_type(16))) _Float16 v16h;
typedef __attribute__((ext_vector_type(8)))  _Float16 v8h;
typedef __attribute__((ext_vector_type(8)))  float    v8f;

__device__ __forceinline__ v16h frag_zero() {
  v16h r;
#pragma unroll
  for (int i = 0; i < 16; ++i) r[i] = (_Float16)0.f;
  return r;
}

__device__ __forceinline__ v8f acc_bcast(float v) {
  v8f r;
#pragma unroll
  for (int j = 0; j < 8; ++j) r[j] = v;
  return r;
}

// 16x32 f16 A/B fragment from LDS; rows contiguous along K.
// pitch in halfs (multiple of 8 -> 16B aligned rows).
__device__ __forceinline__ v16h lds_frag(const _Float16* buf, int pitch, int r0, int k0, int lane) {
  int row = r0 + (lane & 15);
  int off = (lane & 16) ? 8 : 0;
  const _Float16* p = buf + row * pitch + k0 + off;
  v8h lo = *(const v8h*)p;
  v8h hi = *(const v8h*)(p + 16);
  v16h r;
#pragma unroll
  for (int i = 0; i < 8; ++i) { r[i] = lo[i]; r[i + 8] = hi[i]; }
  return r;
}

// same, with row guard (pad rows -> zeros); EXEC restored before WMMA call site
__device__ __forceinline__ v16h lds_frag_g(const _Float16* buf, int pitch, int r0, int k0, int lane, int rlim) {
  int row = r0 + (lane & 15);
  if (row >= rlim) return frag_zero();
  int off = (lane & 16) ? 8 : 0;
  const _Float16* p = buf + row * pitch + k0 + off;
  v8h lo = *(const v8h*)p;
  v8h hi = *(const v8h*)(p + 16);
  v16h r;
#pragma unroll
  for (int i = 0; i < 8; ++i) { r[i] = lo[i]; r[i + 8] = hi[i]; }
  return r;
}

// B fragment from global f16 weight, row-major [N, K] (ldk halfs)
__device__ __forceinline__ v16h glb_frag(const _Float16* w, int ldk, int n0, int k0, int lane) {
  int off = (lane & 16) ? 8 : 0;
  const _Float16* p = w + (size_t)(n0 + (lane & 15)) * ldk + k0 + off;
  v8h lo = *(const v8h*)p;
  v8h hi = *(const v8h*)(p + 16);
  v16h r;
#pragma unroll
  for (int i = 0; i < 8; ++i) { r[i] = lo[i]; r[i + 8] = hi[i]; }
  return r;
}

__device__ __forceinline__ v8f wmma16(v16h a, v16h b, v8f c) {
  return __builtin_amdgcn_wmma_f32_16x16x32_f16(false, a, false, b, (short)0, c, false, false);
}

// ---- CDNA5 async global->LDS copy (ASYNCcnt-tracked), 16B granules ----
__device__ __forceinline__ void async_ldg_b128(unsigned lds_off, const void* gptr) {
  unsigned long long ga = (unsigned long long)(uintptr_t)gptr;
  asm volatile("global_load_async_to_lds_b128 %0, %1, off" :: "v"(lds_off), "v"(ga) : "memory");
}
__device__ __forceinline__ void wait_async0() {
  asm volatile("s_wait_asynccnt 0" ::: "memory");
}

// ---- f32 -> f16 weight staging into workspace ----
// layout (halfs): ipw[0..49152) opw[..65536) w1[..131072) w2[..196608) fe[..199680)
__global__ void prep_f16(const float* __restrict__ ipw, const float* __restrict__ opw,
                         const float* __restrict__ w1, const float* __restrict__ w2,
                         const float* __restrict__ fe, _Float16* __restrict__ ws) {
  int i = blockIdx.x * 256 + threadIdx.x;
  if (i < 49152)       ws[i] = (_Float16)ipw[i];
  else if (i < 65536)  ws[i] = (_Float16)opw[i - 49152];
  else if (i < 131072) ws[i] = (_Float16)w1[i - 65536];
  else if (i < 196608) ws[i] = (_Float16)w2[i - 131072];
  else if (i < 199680) ws[i] = (_Float16)fe[i - 196608];
}

__global__ __launch_bounds__(BD) void rnn_feat_attn(
    const float* __restrict__ x,   const float* __restrict__ Wm,
    const float* __restrict__ Uv,  const float* __restrict__ bvv,
    const float* __restrict__ Am,  const float* __restrict__ cvv,
    const float* __restrict__ z0,
    const float* __restrict__ ipb, const float* __restrict__ opbv,
    const float* __restrict__ ln1g, const float* __restrict__ ln1b,
    const float* __restrict__ b1,  const float* __restrict__ b2,
    const float* __restrict__ ln2g, const float* __restrict__ ln2b,
    const float* __restrict__ opw, const float* __restrict__ opbs,
    const _Float16* __restrict__ w16, float* __restrict__ out)
{
  __shared__ __align__(16) _Float16 s_tok[MROWS * 136];   // tokens f16 (pitch 136)
  __shared__ __align__(16) _Float16 s_qkv[MROWS * 264];   // q|k f16 (256 cols + pad)
  __shared__ __align__(16) _Float16 s_vt [DM * 56];       // V^T: [d][token] (48 + 8 zero pad)
  __shared__ __align__(16) _Float16 s_ao [MROWS * 136];   // attn-out f16
  __shared__ __align__(16) _Float16 s_h  [MROWS * 264];   // ffn hidden chunk f16 (256+pad)
  __shared__ __align__(16) _Float16 s_aw [8 * 32 * 40];   // per-wave attn weights (pitch 40)
  __shared__ __align__(16) _Float16 s_fe [HT * DM];       // feat_embed f16 (async-staged)
  __shared__ __align__(16) _Float16 s_opw[DM * DM];       // out_proj_w f16 (async-staged)
  __shared__ float s_f32[MROWS * 132];                    // f32 scratch (proj / ffn out / scores)
  __shared__ float s_W[HT*HT], s_A[HT*HT], s_U[HT], s_b[HT], s_c[HT];
  __shared__ float s_z[MROWS], s_zn[MROWS];

  const int tid  = threadIdx.x;
  const int lane = tid & 31;
  const int wave = tid >> 5;
  const int hi8  = (lane & 16) ? 8 : 0;
  const int ln   = lane & 15;
  const int b0   = blockIdx.x * NB;

  const _Float16* ipw16 = w16;
  const _Float16* opw16 = w16 + 49152;
  const _Float16* w116  = w16 + 65536;
  const _Float16* w216  = w16 + 131072;
  const _Float16* fe16  = w16 + 196608;

  // stage small weights / state into LDS
  for (int i = tid; i < HT*HT; i += BD) { s_W[i] = Wm[i]; s_A[i] = Am[i]; }
  if (tid < HT) { s_U[tid] = Uv[tid]; s_b[tid] = bvv[tid]; s_c[tid] = cvv[tid]; }
  if (tid < MROWS) s_z[tid] = z0[tid % HT];
  // zero V^T buffer once (pad token columns 48..55 must stay exactly 0)
  for (int i = tid; i < DM * 56; i += BD) s_vt[i] = (_Float16)0.f;
  // async-stage feat_embed (6KB) and out_proj_w (32KB) into LDS
  {
    unsigned fe_lds  = (unsigned)(uintptr_t)(&s_fe[0]);
    unsigned opw_lds = (unsigned)(uintptr_t)(&s_opw[0]);
    for (int i = tid; i < 384; i += BD)               // 24*128 halfs / 8
      async_ldg_b128(fe_lds + i * 16, (const char*)fe16 + i * 16);
    for (int i = tid; i < 2048; i += BD)              // 128*128 halfs / 8
      async_ldg_b128(opw_lds + i * 16, (const char*)opw16 + i * 16);
  }
  // prefetch in_proj, out_proj and w1 (ws halfs [0, 131072)) into L2
  for (int i = tid * 64; i < 131072; i += BD * 64) __builtin_prefetch(ipw16 + i, 0, 1);
  wait_async0();
  __syncthreads();

  const size_t zoff = (size_t)4096 * TT * HT;
  const float scale = 0.17677669529663687f;  // 1/sqrt(32)

  for (int t = 0; t < TT; ++t) {
    // ---------- RNN cell: z = tanh(z@W.T + x*U + b) ----------
    if (tid < MROWS) {
      int bi = tid / HT, h = tid % HT;
      float s = s_b[h] + x[(size_t)(b0 + bi) * TT + t] * s_U[h];
#pragma unroll
      for (int j = 0; j < HT; ++j) s += s_W[h * HT + j] * s_z[bi * HT + j];
      s_zn[tid] = tanhf(s);
    }
    __syncthreads();
    if (tid < MROWS) s_z[tid] = s_zn[tid];
    // tokens = z[:, :, None] * feat_embed
    for (int idx = tid; idx < MROWS * DM; idx += BD) {
      int r = idx >> 7, d = idx & 127;
      s_tok[r * 136 + d] = (_Float16)(s_zn[r] * (float)s_fe[(r % HT) * DM + d]);
    }
    __syncthreads();

    // ---------- QKV = tokens @ in_proj_w.T + b : [48,384], 3x24 tiles ----------
    // q,k columns (n<256) -> s_qkv; v columns (n>=256) -> transposed into s_vt
    for (int tile = wave; tile < 72; tile += 8) {
      int mi = tile % 3, ni = tile / 3, n0 = ni * 16;
      v8f acc = acc_bcast(ipb[n0 + ln]);
#pragma unroll
      for (int kk = 0; kk < 4; ++kk) {
        v16h a  = lds_frag(s_tok, 136, mi * 16, kk * 32, lane);
        v16h bb = glb_frag(ipw16, 128, n0, kk * 32, lane);
        acc = wmma16(a, bb, acc);
      }
      if (n0 < 256) {
#pragma unroll
        for (int j = 0; j < 8; ++j)
          s_qkv[(mi * 16 + j + hi8) * 264 + n0 + ln] = (_Float16)acc[j];
      } else {
        // transposed store: rows m = mi*16+hi8 .. +7 are contiguous -> packed b128
        int dl = n0 - 256 + ln;
        v8h pk;
#pragma unroll
        for (int j = 0; j < 8; ++j) pk[j] = (_Float16)acc[j];
        *(v8h*)(s_vt + dl * 56 + mi * 16 + hi8) = pk;
      }
    }
    __syncthreads();

    // ---------- Attention: one (batch, head) per wave (2*4 = 8 pairs) ----------
    {
      int bp = wave >> 2, hp = wave & 3;
      int rb = bp * HT;
      int qc = hp * HDD, kc = DM + hp * HDD;
      float*    sc = s_f32 + wave * 672;   // 24 x pitch28 scores
      _Float16* aw = s_aw + wave * 1280;   // 32 x pitch40 attn weights
      // scores = q @ k.T * scale
#pragma unroll
      for (int mi = 0; mi < 2; ++mi) {
#pragma unroll
        for (int ni = 0; ni < 2; ++ni) {
          v16h a  = lds_frag_g(s_qkv, 264, rb + mi * 16, qc, lane, rb + HT);
          v16h bb = lds_frag_g(s_qkv, 264, rb + ni * 16, kc, lane, rb + HT);
          v8f acc = acc_bcast(0.f);
          acc = wmma16(a, bb, acc);
#pragma unroll
          for (int j = 0; j < 8; ++j) {
            int m = mi * 16 + j + hi8, n = ni * 16 + ln;
            if (m < HT && n < HT) sc[m * 28 + n] = acc[j] * scale;
          }
        }
      }
      // softmax per row (lane == row)
      if (lane < HT) {
        float mx = -1e30f;
        for (int cI = 0; cI < HT; ++cI) mx = fmaxf(mx, sc[lane * 28 + cI]);
        float ssum = 0.f;
        for (int cI = 0; cI < HT; ++cI) {
          float e = __expf(sc[lane * 28 + cI] - mx);
          sc[lane * 28 + cI] = e; ssum += e;
        }
        float inv = 1.f / ssum;
        for (int cI = 0; cI < HT; ++cI) aw[lane * 40 + cI] = (_Float16)(sc[lane * 28 + cI] * inv);
        for (int cI = HT; cI < 32; ++cI) aw[lane * 40 + cI] = (_Float16)0.f;
      } else {
        for (int cI = 0; cI < 32; ++cI) aw[lane * 40 + cI] = (_Float16)0.f;
      }
      // ao = attw @ v  (B from V^T: row = d, K = token; pad tokens are zeroed)
#pragma unroll
      for (int mi = 0; mi < 2; ++mi) {
#pragma unroll
        for (int ni = 0; ni < 2; ++ni) {
          v16h a  = lds_frag(aw, 40, mi * 16, 0, lane);
          v16h bb = lds_frag(s_vt, 56, hp * HDD + ni * 16, rb, lane);
          v8f acc = acc_bcast(0.f);
          acc = wmma16(a, bb, acc);
#pragma unroll
          for (int j = 0; j < 8; ++j) {
            int m = mi * 16 + j + hi8;
            if (m < HT) s_ao[(rb + m) * 136 + hp * HDD + ni * 16 + ln] = (_Float16)acc[j];
          }
        }
      }
    }
    __syncthreads();

    // ---------- out_proj: [48,128], 3x8 tiles, weights LDS-resident ----------
    for (int tile = wave; tile < 24; tile += 8) {
      int mi = tile % 3, ni = tile / 3, n0 = ni * 16;
      v8f acc = acc_bcast(opbv[n0 + ln]);
#pragma unroll
      for (int kk = 0; kk < 4; ++kk) {
        v16h a  = lds_frag(s_ao, 136, mi * 16, kk * 32, lane);
        v16h bb = lds_frag(s_opw, 128, n0, kk * 32, lane);
        acc = wmma16(a, bb, acc);
      }
#pragma unroll
      for (int j = 0; j < 8; ++j)
        s_f32[(mi * 16 + j + hi8) * 132 + n0 + ln] = acc[j];
    }
    __syncthreads();

    // ---------- LN1 over (tokens + proj) ----------
    if (tid < MROWS) {
      float sm = 0.f, sq = 0.f;
      for (int d = 0; d < DM; ++d) {
        float tv = (float)s_tok[tid * 136 + d] + s_f32[tid * 132 + d];
        sm += tv; sq += tv * tv;
      }
      float mean = sm * (1.f / DM);
      float var  = sq * (1.f / DM) - mean * mean;
      float rs   = rsqrtf(var + 1e-5f);
      for (int d = 0; d < DM; ++d) {
        float tv = (float)s_tok[tid * 136 + d] + s_f32[tid * 132 + d];
        s_tok[tid * 136 + d] = (_Float16)((tv - mean) * rs * ln1g[d] + ln1b[d]);
      }
    }
    __syncthreads();

    // ---------- FFN in two 256-wide chunks ----------
    for (int ch = 0; ch < 2; ++ch) {
      // h = relu(tok @ w1_chunk.T + b1): 3x16 tiles
      for (int tile = wave; tile < 48; tile += 8) {
        int mi = tile % 3, ni = tile / 3;
        int gn0 = ch * 256 + ni * 16;
        v8f acc = acc_bcast(b1[gn0 + ln]);
#pragma unroll
        for (int kk = 0; kk < 4; ++kk) {
          v16h a  = lds_frag(s_tok, 136, mi * 16, kk * 32, lane);
          v16h bb = glb_frag(w116, 128, gn0, kk * 32, lane);
          acc = wmma16(a, bb, acc);
        }
#pragma unroll
        for (int j = 0; j < 8; ++j)
          s_h[(mi * 16 + j + hi8) * 264 + ni * 16 + ln] = (_Float16)fmaxf(acc[j], 0.f);
      }
      __syncthreads();
      // out (+=) h @ w2_chunk.T : 3x8 tiles, K = 256 of this chunk
      for (int tile = wave; tile < 24; tile += 8) {
        int mi = tile % 3, ni = tile / 3, n0 = ni * 16;
        v8f acc;
        if (ch == 0) acc = acc_bcast(b2[n0 + ln]);
        else {
#pragma unroll
          for (int j = 0; j < 8; ++j) acc[j] = s_f32[(mi * 16 + j + hi8) * 132 + n0 + ln];
        }
#pragma unroll
        for (int kk = 0; kk < 8; ++kk) {
          v16h a  = lds_frag(s_h, 264, mi * 16, kk * 32, lane);
          v16h bb = glb_frag(w216, 512, n0, ch * 256 + kk * 32, lane);
          acc = wmma16(a, bb, acc);
        }
#pragma unroll
        for (int j = 0; j < 8; ++j)
          s_f32[(mi * 16 + j + hi8) * 132 + n0 + ln] = acc[j];
      }
      __syncthreads();
    }

    // ---------- LN2 + delta = tokens2 @ op_w.T ; z += delta ----------
    if (tid < MROWS) {
      float sm = 0.f, sq = 0.f;
      for (int d = 0; d < DM; ++d) {
        float tv = (float)s_tok[tid * 136 + d] + s_f32[tid * 132 + d];
        sm += tv; sq += tv * tv;
      }
      float mean = sm * (1.f / DM);
      float var  = sq * (1.f / DM) - mean * mean;
      float rs   = rsqrtf(var + 1e-5f);
      float delta = opbs[0];
      for (int d = 0; d < DM; ++d) {
        float tv = (float)s_tok[tid * 136 + d] + s_f32[tid * 132 + d];
        float y  = (tv - mean) * rs * ln2g[d] + ln2b[d];
        delta += y * opw[d];
      }
      s_z[tid] += delta;
    }
    __syncthreads();

    // ---------- outputs: o = z @ A.T + c ; store O and Z ----------
    if (tid < MROWS) {
      int bi = tid / HT, h = tid % HT;
      float o = s_c[h];
#pragma unroll
      for (int j = 0; j < HT; ++j) o += s_A[h * HT + j] * s_z[bi * HT + j];
      size_t ob = ((size_t)(b0 + bi) * TT + t) * HT + h;
      out[ob] = o;
      out[zoff + ob] = s_z[tid];
    }
    __syncthreads();
  }
}

extern "C" void kernel_launch(void* const* d_in, const int* in_sizes, int n_in,
                              void* d_out, int out_size, void* d_ws, size_t ws_size,
                              hipStream_t stream) {
  (void)in_sizes; (void)n_in; (void)out_size; (void)ws_size;
  const float* x    = (const float*)d_in[0];
  const float* W    = (const float*)d_in[1];
  const float* U    = (const float*)d_in[2];
  const float* b    = (const float*)d_in[3];
  const float* A    = (const float*)d_in[4];
  const float* c    = (const float*)d_in[5];
  const float* z0   = (const float*)d_in[6];
  const float* fe   = (const float*)d_in[7];
  const float* ipw  = (const float*)d_in[8];
  const float* ipb  = (const float*)d_in[9];
  const float* opw  = (const float*)d_in[10];
  const float* opb  = (const float*)d_in[11];
  const float* ln1g = (const float*)d_in[12];
  const float* ln1b = (const float*)d_in[13];
  const float* w1   = (const float*)d_in[14];
  const float* b1   = (const float*)d_in[15];
  const float* w2   = (const float*)d_in[16];
  const float* b2   = (const float*)d_in[17];
  const float* ln2g = (const float*)d_in[18];
  const float* ln2b = (const float*)d_in[19];
  const float* opw1 = (const float*)d_in[20];
  const float* opb1 = (const float*)d_in[21];

  _Float16* w16 = (_Float16*)d_ws;
  prep_f16<<<780, 256, 0, stream>>>(ipw, opw, w1, w2, fe, w16);
  rnn_feat_attn<<<4096 / NB, BD, 0, stream>>>(
      x, W, U, b, A, c, z0, ipb, opb, ln1g, ln1b, b1, b2, ln2g, ln2b,
      opw1, opb1, w16, (float*)d_out);
}
